// SparsePropagation_26216480375150
// MI455X (gfx1250) — compile-verified, matrix operand order checked by reading, not run on hardware
//
#include <hip/hip_runtime.h>

typedef __attribute__((ext_vector_type(16))) __bf16 v16bf;
typedef __attribute__((ext_vector_type(8)))  __bf16 v8bf;
typedef __attribute__((ext_vector_type(2)))  __bf16 v2bf;
typedef __attribute__((ext_vector_type(8)))  float  v8f;

#define B_       4
#define N_       4096
#define D_       256
#define TOPK_    64
#define ROWS     16
#define NTHREADS 256
#define NWAVES   8
#define SCALE_   0.0625f   /* 1/sqrt(256) */
#define SSTRIDE  (N_ + 16)  /* padded score row stride (dwords): 4112 % 64 = 16 -> no bank conflict */

// LDS budget (dynamic): scores 16*4112*4 = 263168
//                       aux (hist u32 / 2x staged val^T bf16 buffers) 32768
//                       thr/dstate/prefix/krem 4*16*4 = 256
//                       A tile bf16 16*256*2 = 8192    => 304,384 B (< 320 KB)
#define SMEM_SCORES   0
#define SMEM_AUX      (ROWS*SSTRIDE*4)
#define SMEM_THR      (SMEM_AUX + 2*ROWS*256*4)
#define SMEM_DST      (SMEM_THR + ROWS*4)
#define SMEM_PREFIX   (SMEM_DST + ROWS*4)
#define SMEM_KREM     (SMEM_PREFIX + ROWS*4)
#define SMEM_ATILE    (SMEM_KREM + ROWS*4)
#define SMEM_TOTAL    (SMEM_ATILE + ROWS*D_*2)

__global__ void __launch_bounds__(NTHREADS)
sparse_prop_fused(const float* __restrict__ val, const float* __restrict__ state,
                  float* __restrict__ out_state, float* __restrict__ out_val)
{
    extern __shared__ unsigned char smem[];
    float*    sScores = (float*)(smem + SMEM_SCORES);      // [ROWS][SSTRIDE]
    float*    sAux    = (float*)(smem + SMEM_AUX);         // hist(u32) then 2 staged tiles
    float*    sThr    = (float*)(smem + SMEM_THR);         // [ROWS]
    float*    sDSt    = (float*)(smem + SMEM_DST);         // [ROWS]
    unsigned* sPref   = (unsigned*)(smem + SMEM_PREFIX);   // [ROWS]
    unsigned* sKRem   = (unsigned*)(smem + SMEM_KREM);     // [ROWS]
    __bf16*   sA      = (__bf16*)(smem + SMEM_ATILE);      // [ROWS][D_]

    const int tid   = threadIdx.x;
    const int wave  = tid >> 5;
    const int lane  = tid & 31;
    const int nlo   = lane & 15;           // M (A/E) or N (B/C) index
    const int khalf = (lane >> 4) << 3;    // 0 / 8 : K sub-block per A/B layout
    const int mBase = (lane >> 4) << 3;    // 0 / 8 : C/D row base per layout

    const int blk = blockIdx.x;
    const int b   = blk / (N_ / ROWS);
    const int i0  = (blk % (N_ / ROWS)) * ROWS;

    const float* valB = val   + (size_t)b * N_ * D_;
    const float* stB  = state + (size_t)b * N_;

    // ---- Phase 0: stage A tile (16 x 256) as bf16 in LDS ----
    #pragma unroll
    for (int k = 0; k < (ROWS * D_) / NTHREADS; ++k) {
        int idx = tid + k * NTHREADS;
        sA[idx] = (__bf16)valB[(size_t)i0 * D_ + idx];
    }
    __syncthreads();

    // A fragments: 8 K-steps of 16x32 bf16.
    // Lanes 0-15: row=lane, K = {kk*32+0..7, kk*32+16..23}; lanes 16-31: K shifted by 8.
    v16bf afrag[8];
    #pragma unroll
    for (int kk = 0; kk < 8; ++kk) {
        const __bf16* src = sA + nlo * D_ + kk * 32 + khalf;
        v8bf lo = *(const v8bf*)(src);
        v8bf hi = *(const v8bf*)(src + 16);
        afrag[kk] = __builtin_shufflevector(lo, hi, 0,1,2,3,4,5,6,7,8,9,10,11,12,13,14,15);
    }

    // ---- Phase 1: scores(16 x 4096) = A . val^T * SCALE, via WMMA bf16 ----
    for (int jt = wave; jt < N_ / 16; jt += NWAVES) {
        const int j0 = jt * 16;
        const float* brow = valB + (size_t)(j0 + nlo) * D_;   // B col n = val row j0+n
        v8f c = {};
        #pragma unroll
        for (int kk = 0; kk < 8; ++kk) {
            const int base = kk * 32 + khalf;
            float4 a0 = *(const float4*)(brow + base);
            float4 a1 = *(const float4*)(brow + base + 4);
            float4 b0 = *(const float4*)(brow + base + 16);
            float4 b1 = *(const float4*)(brow + base + 20);
            v16bf bf;
            bf[0]=(__bf16)a0.x;  bf[1]=(__bf16)a0.y;  bf[2]=(__bf16)a0.z;  bf[3]=(__bf16)a0.w;
            bf[4]=(__bf16)a1.x;  bf[5]=(__bf16)a1.y;  bf[6]=(__bf16)a1.z;  bf[7]=(__bf16)a1.w;
            bf[8]=(__bf16)b0.x;  bf[9]=(__bf16)b0.y;  bf[10]=(__bf16)b0.z; bf[11]=(__bf16)b0.w;
            bf[12]=(__bf16)b1.x; bf[13]=(__bf16)b1.y; bf[14]=(__bf16)b1.z; bf[15]=(__bf16)b1.w;
            c = __builtin_amdgcn_wmma_f32_16x16x32_bf16(false, afrag[kk], false, bf,
                                                        (short)0, c, false, false);
        }
        #pragma unroll
        for (int r = 0; r < 8; ++r)
            sScores[(mBase + r) * SSTRIDE + j0 + nlo] = c[r] * SCALE_;
    }
    __syncthreads();

    // ---- Phase 2: exact k-th largest per row (radix-256 select, 4 passes) ----
    const int row = tid >> 4;        // 16 threads per row
    const int tr  = tid & 15;
    unsigned* hist = (unsigned*)sAux;

    if (tr == 0) { sPref[row] = 0u; sKRem[row] = (unsigned)TOPK_; }
    __syncthreads();

    for (int pass = 3; pass >= 0; --pass) {
        #pragma unroll
        for (int k = 0; k < 16; ++k) hist[tid + k * NTHREADS] = 0u;
        __syncthreads();

        const unsigned msk  = (pass == 3) ? 0u : (0xFFFFFFFFu << ((pass + 1) * 8));
        const unsigned pref = sPref[row];
        for (int c = tr * (N_ / 16); c < (tr + 1) * (N_ / 16); ++c) {
            unsigned u   = __float_as_uint(sScores[row * SSTRIDE + c]);
            unsigned key = (u & 0x80000000u) ? ~u : (u | 0x80000000u);  // order-preserving
            if ((key & msk) == pref)
                atomicAdd(&hist[(row << 8) + ((key >> (pass * 8)) & 0xFFu)], 1u);
        }
        __syncthreads();

        if (tr == 0) {
            unsigned kRem = sKRem[row], cum = 0; int sel = 0;
            for (int bin = 255; bin >= 0; --bin) {
                unsigned cnt = hist[(row << 8) + bin];
                if (cum + cnt >= kRem) { sel = bin; break; }
                cum += cnt;
            }
            sPref[row] = pref | ((unsigned)sel << (pass * 8));
            sKRem[row] = kRem - cum;
        }
        __syncthreads();
    }

    if (tr == 0) {
        unsigned k = sPref[row];
        unsigned u = (k & 0x80000000u) ? (k ^ 0x80000000u) : ~k;
        sThr[row] = __uint_as_float(u);      // exact k-th largest score of this row
        sDSt[row] = 0.0f;
    }
    __syncthreads();

    // delta_state[i] = sum_j softsign(s)*(s>=thr) * state[j]
    {
        const float thr = sThr[row];
        float acc = 0.0f;
        for (int c = tr * (N_ / 16); c < (tr + 1) * (N_ / 16); ++c) {
            float s = sScores[row * SSTRIDE + c];
            if (s >= thr) acc += (s / (1.0f + fabsf(s))) * stB[c];
        }
        atomicAdd(&sDSt[row], acc);
    }

    // ---- Phase 3: delta_val(16 x 256) = E(16 x 4096) . val(4096 x 256), WMMA bf16 ----
    // All waves march the same 32-wide j-step. val[j0:j0+32,:] is cooperatively staged
    // into LDS transposed as bf16 (sBT[d][k]) with DOUBLE BUFFERING: one barrier per
    // step, next step's global loads overlap current step's WMMA/DS work. Each wave
    // owns 2 distinct D-tiles; accumulators live in VGPRs and go straight to global.
    __bf16* sBT0 = (__bf16*)sAux;              // 2 x [D_][32] bf16 buffers (16 KB each)
    const float thr_m = sThr[nlo];
    const int   dt0   = wave * 2;              // this wave's first D-tile
    v8f acc0 = {}, acc1 = {};

    const int k2 = (tid & 15) * 2;             // staging: rows j0+k2, j0+k2+1
    const int dc = (tid >> 4) * 16;            // staging: 16 d-columns

    auto stage = [&](int js, int buf) {
        __bf16* sBT = sBT0 + (size_t)buf * (D_ * 32);
        const float* r0 = valB + (size_t)(js * 32 + k2) * D_ + dc;
        const float* r1 = r0 + D_;
        #pragma unroll
        for (int q = 0; q < 16; q += 4) {
            float4 x0 = *(const float4*)(r0 + q);
            float4 x1 = *(const float4*)(r1 + q);
            v2bf p0; p0[0] = (__bf16)x0.x; p0[1] = (__bf16)x1.x;
            v2bf p1; p1[0] = (__bf16)x0.y; p1[1] = (__bf16)x1.y;
            v2bf p2; p2[0] = (__bf16)x0.z; p2[1] = (__bf16)x1.z;
            v2bf p3; p3[0] = (__bf16)x0.w; p3[1] = (__bf16)x1.w;
            *(v2bf*)(sBT + (size_t)(dc + q + 0) * 32 + k2) = p0;
            *(v2bf*)(sBT + (size_t)(dc + q + 1) * 32 + k2) = p1;
            *(v2bf*)(sBT + (size_t)(dc + q + 2) * 32 + k2) = p2;
            *(v2bf*)(sBT + (size_t)(dc + q + 3) * 32 + k2) = p3;
        }
    };

    stage(0, 0);                               // prologue (hist region is dead past the
                                               // barrier above, so this is race-free)
    for (int js = 0; js < N_ / 32; ++js) {
        __syncthreads();                       // buf[js&1] staged; buf[(js+1)&1] reads done
        if (js + 1 < N_ / 32) stage(js + 1, (js + 1) & 1);

        const int j0 = js * 32;
        __bf16* sBT = sBT0 + (size_t)(js & 1) * (D_ * 32);

        // E fragment (masked softsign of scores) from LDS
        v16bf e;
        const float* srow = sScores + nlo * SSTRIDE + j0 + khalf;
        #pragma unroll
        for (int q = 0; q < 8; ++q) {
            float s = srow[q];
            e[q] = (__bf16)((s >= thr_m) ? (s / (1.0f + fabsf(s))) : 0.0f);
        }
        #pragma unroll
        for (int q = 0; q < 8; ++q) {
            float s = srow[16 + q];
            e[8 + q] = (__bf16)((s >= thr_m) ? (s / (1.0f + fabsf(s))) : 0.0f);
        }

        // B fragments: 2 x ds_load_b128 each from transposed staged tile
        {
            const __bf16* bp = sBT + (size_t)(dt0 * 16 + nlo) * 32 + khalf;
            v8bf blo = *(const v8bf*)(bp);
            v8bf bhi = *(const v8bf*)(bp + 16);
            v16bf bf0 = __builtin_shufflevector(blo, bhi, 0,1,2,3,4,5,6,7,8,9,10,11,12,13,14,15);
            acc0 = __builtin_amdgcn_wmma_f32_16x16x32_bf16(false, e, false, bf0,
                                                           (short)0, acc0, false, false);
        }
        {
            const __bf16* bp = sBT + (size_t)((dt0 + 1) * 16 + nlo) * 32 + khalf;
            v8bf blo = *(const v8bf*)(bp);
            v8bf bhi = *(const v8bf*)(bp + 16);
            v16bf bf1 = __builtin_shufflevector(blo, bhi, 0,1,2,3,4,5,6,7,8,9,10,11,12,13,14,15);
            acc1 = __builtin_amdgcn_wmma_f32_16x16x32_bf16(false, e, false, bf1,
                                                           (short)0, acc1, false, false);
        }
    }

    // ---- Outputs (delta_val straight from accumulators; C layout: row mBase+r, col = nlo) ----
    {
        float* dst0 = out_val + ((size_t)b * N_ + i0 + mBase) * D_ + dt0 * 16 + nlo;
        #pragma unroll
        for (int r = 0; r < 8; ++r) {
            dst0[(size_t)r * D_]      = acc0[r];
            dst0[(size_t)r * D_ + 16] = acc1[r];
        }
    }
    if (tid < ROWS) out_state[(size_t)b * N_ + i0 + tid] = sDSt[tid];
}

extern "C" void kernel_launch(void* const* d_in, const int* in_sizes, int n_in,
                              void* d_out, int out_size, void* d_ws, size_t ws_size,
                              hipStream_t stream) {
    const float* val   = (const float*)d_in[0];   // [B, N, D] f32
    const float* state = (const float*)d_in[1];   // [B, N]    f32
    float* out_state = (float*)d_out;                         // [B, N]
    float* out_val   = (float*)d_out + (size_t)B_ * N_;       // [B, N, D]

    const size_t smem = (size_t)SMEM_TOTAL;       // ~304 KB, fits CDNA5's 320 KB WGP LDS
    (void)hipFuncSetAttribute((const void*)sparse_prop_fused,
                              hipFuncAttributeMaxDynamicSharedMemorySize, (int)smem);

    dim3 grid(B_ * (N_ / ROWS));   // 1024 workgroups
    dim3 block(NTHREADS);          // 8 wave32
    sparse_prop_fused<<<grid, block, smem, stream>>>(val, state, out_state, out_val);
}
